// TransformerBlock_24756191494437
// MI455X (gfx1250) — matrix operand order repaired and checked by reading, not compiled, not run on hardware
//
#include <hip/hip_runtime.h>
#include <hip/hip_bf16.h>

typedef __bf16 bf16_t;
typedef __attribute__((ext_vector_type(4)))  bf16_t v4bf;
typedef __attribute__((ext_vector_type(8)))  bf16_t v8bf;
typedef __attribute__((ext_vector_type(16))) bf16_t v16bf;
typedef __attribute__((ext_vector_type(8)))  float  v8f;

#define BATCH 1024
#define PP 9
#define DD 2700
#define DFFC 512
#define MROWS (BATCH * PP)   // 9216
#define KAUX (2 * DFFC)      // 1024

#define BM 128
#define BN 128
#define BK 64
#define SA 72                // LDS row stride in bf16 elems (pad 8) -> 144B rows, 16B aligned

// ---------------- conversion kernels ----------------
__global__ void cvt_f32_to_bf16(const float* __restrict__ in, bf16_t* __restrict__ out, int n) {
    int i = blockIdx.x * blockDim.x + threadIdx.x;
    int stride = gridDim.x * blockDim.x;
    for (; i < n; i += stride) out[i] = (bf16_t)in[i];
}

// out: rows x 1024 (bf16) = [a | b] with a,b rows x 512 (f32)
__global__ void concat_rows_bf16(const float* __restrict__ a, const float* __restrict__ b,
                                 bf16_t* __restrict__ out, int rows) {
    int i = blockIdx.x * blockDim.x + threadIdx.x;
    int stride = gridDim.x * blockDim.x;
    int n = rows * 1024;
    for (; i < n; i += stride) {
        int r = i >> 10, c = i & 1023;
        float v = (c < 512) ? a[r * 512 + c] : b[r * 512 + (c - 512)];
        out[i] = (bf16_t)v;
    }
}

// out[(c)*ldout + co + r] = bf16(in[r*C + c])   (transpose + convert, 32x32 LDS tiles)
__global__ __launch_bounds__(256)
void transpose_cvt_kernel(const float* __restrict__ in, bf16_t* __restrict__ out,
                          int R, int C, int ldout, int co) {
    __shared__ float tile[32][33];
    const int bx = blockIdx.x * 32;  // C offset
    const int by = blockIdx.y * 32;  // R offset
    const int tx = threadIdx.x & 31;
    const int ty = threadIdx.x >> 5;  // 0..7
#pragma unroll
    for (int i = 0; i < 4; ++i) {
        int r = by + ty + i * 8;
        int c = bx + tx;
        tile[ty + i * 8][tx] = (r < R && c < C) ? in[(size_t)r * C + c] : 0.f;
    }
    __syncthreads();
#pragma unroll
    for (int i = 0; i < 4; ++i) {
        int cc = bx + ty + i * 8;  // output row (= input column)
        int rr = by + tx;          // output col (= input row)
        if (cc < C && rr < R)
            out[(size_t)cc * ldout + co + rr] = (bf16_t)tile[tx][ty + i * 8];
    }
}

// ---------------- fused modulated-projection GEMM ----------------
// Both operands K-contiguous: A = rows x KLEN (row-major), Bt = cols(N) x KLEN (row-major).
// Rows are CLAMPED (not predicated): out-of-range B rows feed columns the epilogue discards.
// FULLK=true -> branch-free loads; FULLK=false only for the single partial K-step.
template <int KLEN, bool FULLK>
__device__ inline void stage_regs(const bf16_t* __restrict__ P, int r0, int rlim, int kk,
                                  int t, v4bf (&regs)[8]) {
#pragma unroll
    for (int p = 0; p < 8; ++p) {
        int row = r0 + p * 16 + (t >> 4);
        row = row < rlim ? row : (rlim - 1);
        int col = (t & 15) * 4;
        if (FULLK || (kk + col + 4) <= KLEN) {
            regs[p] = *(const v4bf*)(P + (size_t)row * KLEN + kk + col);
        } else {
            regs[p][0] = (bf16_t)0.f; regs[p][1] = (bf16_t)0.f;
            regs[p][2] = (bf16_t)0.f; regs[p][3] = (bf16_t)0.f;
        }
    }
}

__device__ inline void store_lds(bf16_t* S, int t, const v4bf (&regs)[8]) {
#pragma unroll
    for (int p = 0; p < 8; ++p) {
        int row = p * 16 + (t >> 4);
        int col = (t & 15) * 4;
        *(v4bf*)(&S[row * SA + col]) = regs[p];
    }
}

template <int KLEN>
__device__ inline void gemm_loop(const bf16_t* __restrict__ Amat,  // M x KLEN
                                 const bf16_t* __restrict__ Bt,    // DD x KLEN (N-major)
                                 int m0, int n0,
                                 bf16_t* As, bf16_t* Bs,
                                 int t, int mwarp, int nwarp, int half, int lm,
                                 v8f (&acc)[2][4]) {
    v4bf ar[8], br[8];
    stage_regs<KLEN, true>(Amat, m0, MROWS, 0, t, ar);
    stage_regs<KLEN, true>(Bt,   n0, DD,    0, t, br);

    for (int kk = 0; kk < KLEN; kk += BK) {
        __syncthreads();
        store_lds(As, t, ar);
        store_lds(Bs, t, br);
        __syncthreads();
        // software pipeline: issue next tile's global loads before compute
        int kn = kk + BK;
        if (kn < KLEN) {
            if (kn + BK <= KLEN) {
                stage_regs<KLEN, true>(Amat, m0, MROWS, kn, t, ar);
                stage_regs<KLEN, true>(Bt,   n0, DD,    kn, t, br);
            } else {
                stage_regs<KLEN, false>(Amat, m0, MROWS, kn, t, ar);
                stage_regs<KLEN, false>(Bt,   n0, DD,    kn, t, br);
            }
        }
        // ---- fragments + WMMA: two 32-wide K sub-steps per staged 64-wide tile ----
#pragma unroll
        for (int ks = 0; ks < 2; ++ks) {
            const int ko = ks * 32;
            v16bf bfr[4];
#pragma unroll
            for (int nt = 0; nt < 4; ++nt) {
                // B layout: lane (n = lane&15), half 0 -> K 0..15, half 1 -> K 16..31
                const bf16_t* rp = &Bs[(nwarp + nt * 16 + lm) * SA + ko];
                v8bf lo = *(const v8bf*)(rp + half * 16);
                v8bf hi = *(const v8bf*)(rp + half * 16 + 8);
                bfr[nt] = __builtin_shufflevector(lo, hi, 0, 1, 2, 3, 4, 5, 6, 7,
                                                  8, 9, 10, 11, 12, 13, 14, 15);
            }
#pragma unroll
            for (int mt = 0; mt < 2; ++mt) {
                // A layout: lane (m = lane&15), half 0 -> K 0..15, half 1 -> K 16..31
                const bf16_t* rp = &As[(mwarp + mt * 16 + lm) * SA + ko];
                v8bf lo = *(const v8bf*)(rp + half * 16);
                v8bf hi = *(const v8bf*)(rp + half * 16 + 8);
                v16bf afr = __builtin_shufflevector(lo, hi, 0, 1, 2, 3, 4, 5, 6, 7,
                                                    8, 9, 10, 11, 12, 13, 14, 15);
#pragma unroll
                for (int nt = 0; nt < 4; ++nt) {
                    acc[mt][nt] = __builtin_amdgcn_wmma_f32_16x16x32_bf16(
                        false, afr, false, bfr[nt], (short)0, acc[mt][nt], false, false);
                }
            }
        }
    }
}

// Out = (X@W + bmain) * (H@WC + baux1 + baux2), stored bf16. W/WC passed pre-transposed (N x K).
__global__ __launch_bounds__(256)
void gemm_mod_kernel(const bf16_t* __restrict__ X,    // MROWS x DD
                     const bf16_t* __restrict__ Wt,   // DD x DD   (N-major)
                     const float*  __restrict__ bmain,
                     const bf16_t* __restrict__ H,    // MROWS x KAUX
                     const bf16_t* __restrict__ WCt,  // DD x KAUX (N-major)
                     const float*  __restrict__ baux1,
                     const float*  __restrict__ baux2,
                     bf16_t* __restrict__ Out)        // MROWS x DD
{
    __shared__ bf16_t As[BM * SA];
    __shared__ bf16_t Bs[BN * SA];

    const int t    = threadIdx.x;
    const int lane = t & 31;
    const int wave = t >> 5;
    const int half = lane >> 4;
    const int lm   = lane & 15;
    const int mwarp = (wave & 3) * 32;   // 4 wave-rows of 32
    const int nwarp = (wave >> 2) * 64;  // 2 wave-cols of 64
    const int m0 = blockIdx.y * BM;
    const int n0 = blockIdx.x * BN;

    v8f acc1[2][4], acc2[2][4];
#pragma unroll
    for (int i = 0; i < 2; ++i)
#pragma unroll
        for (int j = 0; j < 4; ++j)
#pragma unroll
            for (int r = 0; r < 8; ++r) { acc1[i][j][r] = 0.f; acc2[i][j][r] = 0.f; }

    gemm_loop<DD>(X, Wt, m0, n0, As, Bs, t, mwarp, nwarp, half, lm, acc1);
    gemm_loop<KAUX>(H, WCt, m0, n0, As, Bs, t, mwarp, nwarp, half, lm, acc2);

    // epilogue: C/D layout -> element r of lane l: m = r + 8*half, n = lane&15
#pragma unroll
    for (int mt = 0; mt < 2; ++mt)
#pragma unroll
        for (int nt = 0; nt < 4; ++nt) {
            int gn = n0 + nwarp + nt * 16 + lm;
            if (gn < DD) {
                float b0 = bmain[gn];
                float b1 = baux1[gn] + baux2[gn];
#pragma unroll
                for (int r = 0; r < 8; ++r) {
                    int gm = m0 + mwarp + mt * 16 + r + 8 * half;
                    float v = (acc1[mt][nt][r] + b0) * (acc2[mt][nt][r] + b1);
                    Out[(size_t)gm * DD + gn] = (bf16_t)v;
                }
            }
        }
}

// ---------------- attention scores + softmax ----------------
__global__ __launch_bounds__(256)
void attn_scores_kernel(const bf16_t* __restrict__ Q, const bf16_t* __restrict__ Km,
                        float* __restrict__ Aout) {
    const int ri = blockIdx.x;      // (b*9 + q) row
    const int b  = ri / PP;
    const int t  = threadIdx.x;

    float part[PP];
#pragma unroll
    for (int j = 0; j < PP; ++j) part[j] = 0.f;

    const bf16_t* qrow  = Q + (size_t)ri * DD;
    const bf16_t* kbase = Km + (size_t)b * PP * DD;
    for (int d = t; d < DD; d += 256) {
        float qv = (float)qrow[d];
#pragma unroll
        for (int j = 0; j < PP; ++j)
            part[j] += qv * (float)kbase[(size_t)j * DD + d];
    }

    __shared__ float ssum[PP];
    if (t < PP) ssum[t] = 0.f;
    __syncthreads();
#pragma unroll
    for (int j = 0; j < PP; ++j) {
        float v = part[j];
        for (int off = 16; off > 0; off >>= 1) v += __shfl_down(v, off, 32);
        if ((t & 31) == 0) atomicAdd(&ssum[j], v);
    }
    __syncthreads();
    if (t == 0) {
        const float scale = rsqrtf((float)DD);
        float s[PP], mx = -1e30f, sum = 0.f;
#pragma unroll
        for (int j = 0; j < PP; ++j) { s[j] = ssum[j] * scale; mx = fmaxf(mx, s[j]); }
#pragma unroll
        for (int j = 0; j < PP; ++j) { s[j] = expf(s[j] - mx); sum += s[j]; }
        float inv = 1.f / sum;
#pragma unroll
        for (int j = 0; j < PP; ++j) Aout[(size_t)ri * PP + j] = s[j] * inv;
    }
}

// ---------------- attn@V + residual + LayerNorm ----------------
__global__ __launch_bounds__(256)
void attn_out_ln_kernel(const float* __restrict__ Xf, const bf16_t* __restrict__ V,
                        const float* __restrict__ Aw,
                        const float* __restrict__ gamma, const float* __restrict__ beta,
                        float* __restrict__ Out) {
    const int ri = blockIdx.x;
    const int b  = ri / PP;
    const int t  = threadIdx.x;

    float aw[PP];
#pragma unroll
    for (int j = 0; j < PP; ++j) aw[j] = Aw[(size_t)ri * PP + j];

    const float*  xrow  = Xf + (size_t)ri * DD;
    const bf16_t* vbase = V + (size_t)b * PP * DD;

    float z[11];
    float sum = 0.f, sq = 0.f;
#pragma unroll
    for (int it = 0; it < 11; ++it) {
        int d = t + it * 256;
        float acc = 0.f;
        if (d < DD) {
            acc = xrow[d];
#pragma unroll
            for (int j = 0; j < PP; ++j) acc += aw[j] * (float)vbase[(size_t)j * DD + d];
            sum += acc; sq += acc * acc;
        }
        z[it] = acc;
    }

    for (int off = 16; off > 0; off >>= 1) {
        sum += __shfl_down(sum, off, 32);
        sq  += __shfl_down(sq,  off, 32);
    }
    __shared__ float s1[8], s2[8];
    __shared__ float smean, srinv;
    if ((t & 31) == 0) { s1[t >> 5] = sum; s2[t >> 5] = sq; }
    __syncthreads();
    if (t == 0) {
        float ts = 0.f, tq = 0.f;
        for (int w = 0; w < 8; ++w) { ts += s1[w]; tq += s2[w]; }
        float mean = ts / (float)DD;
        float var  = tq / (float)DD - mean * mean;
        smean = mean;
        srinv = rsqrtf(var + 1e-5f);
    }
    __syncthreads();
    float mean = smean, rinv = srinv;
    float* orow = Out + (size_t)ri * DD;
#pragma unroll
    for (int it = 0; it < 11; ++it) {
        int d = t + it * 256;
        if (d < DD) orow[d] = (z[it] - mean) * rinv * gamma[d] + beta[d];
    }
}

// ---------------- host launcher ----------------
extern "C" void kernel_launch(void* const* d_in, const int* in_sizes, int n_in,
                              void* d_out, int out_size, void* d_ws, size_t ws_size,
                              hipStream_t stream) {
    (void)in_sizes; (void)n_in; (void)out_size; (void)ws_size;
    const float* state = (const float*)d_in[0];
    const float* H1    = (const float*)d_in[1];
    const float* H2    = (const float*)d_in[2];
    const float* Wq    = (const float*)d_in[3];  const float* bq   = (const float*)d_in[4];
    const float* Wk    = (const float*)d_in[5];  const float* bk   = (const float*)d_in[6];
    const float* Wv    = (const float*)d_in[7];  const float* bv   = (const float*)d_in[8];
    const float* WC1q  = (const float*)d_in[9];  const float* bC1q = (const float*)d_in[10];
    const float* WC1k  = (const float*)d_in[11]; const float* bC1k = (const float*)d_in[12];
    const float* WC1v  = (const float*)d_in[13]; const float* bC1v = (const float*)d_in[14];
    const float* WC2q  = (const float*)d_in[15]; const float* bC2q = (const float*)d_in[16];
    const float* WC2k  = (const float*)d_in[17]; const float* bC2k = (const float*)d_in[18];
    const float* WC2v  = (const float*)d_in[19]; const float* bC2v = (const float*)d_in[20];
    const float* gamma = (const float*)d_in[21]; const float* beta = (const float*)d_in[22];
    float* out = (float*)d_out;

    char* ws = (char*)d_ws;
    size_t off = 0;
    auto alloc = [&](size_t bytes) {
        void* p = ws + off;
        off += (bytes + 255) & ~(size_t)255;
        return p;
    };
    bf16_t* xb   = (bf16_t*)alloc((size_t)MROWS * DD * 2);
    bf16_t* Hcat = (bf16_t*)alloc((size_t)MROWS * KAUX * 2);
    bf16_t* Wqt  = (bf16_t*)alloc((size_t)DD * DD * 2);     // transposed: N x K
    bf16_t* Wkt  = (bf16_t*)alloc((size_t)DD * DD * 2);
    bf16_t* Wvt  = (bf16_t*)alloc((size_t)DD * DD * 2);
    bf16_t* WCqt = (bf16_t*)alloc((size_t)DD * KAUX * 2);   // transposed: N x K2
    bf16_t* WCkt = (bf16_t*)alloc((size_t)DD * KAUX * 2);
    bf16_t* WCvt = (bf16_t*)alloc((size_t)DD * KAUX * 2);
    bf16_t* qb   = (bf16_t*)alloc((size_t)MROWS * DD * 2);
    bf16_t* kb   = (bf16_t*)alloc((size_t)MROWS * DD * 2);
    bf16_t* vb   = (bf16_t*)alloc((size_t)MROWS * DD * 2);
    float*  Aws  = (float*)alloc((size_t)MROWS * PP * 4);

    auto cdiv = [](int a, int b) { return (a + b - 1) / b; };
    const int thr = 256;
    const int nX  = MROWS * DD;  // 24,883,200

    // 1) conversions (+ weight transposes to N-major bf16)
    cvt_f32_to_bf16<<<cdiv(nX, thr * 8), thr, 0, stream>>>(state, xb, nX);
    concat_rows_bf16<<<cdiv(MROWS * KAUX, thr * 8), thr, 0, stream>>>(H1, H2, Hcat, MROWS);

    dim3 tgridW(cdiv(DD, 32), cdiv(DD, 32));     // W: R=2700, C=2700
    transpose_cvt_kernel<<<tgridW, thr, 0, stream>>>(Wq, Wqt, DD, DD, DD, 0);
    transpose_cvt_kernel<<<tgridW, thr, 0, stream>>>(Wk, Wkt, DD, DD, DD, 0);
    transpose_cvt_kernel<<<tgridW, thr, 0, stream>>>(Wv, Wvt, DD, DD, DD, 0);

    dim3 tgridC(cdiv(DD, 32), cdiv(DFFC, 32));   // WC half: R=512, C=2700
    transpose_cvt_kernel<<<tgridC, thr, 0, stream>>>(WC1q, WCqt, DFFC, DD, KAUX, 0);
    transpose_cvt_kernel<<<tgridC, thr, 0, stream>>>(WC2q, WCqt, DFFC, DD, KAUX, DFFC);
    transpose_cvt_kernel<<<tgridC, thr, 0, stream>>>(WC1k, WCkt, DFFC, DD, KAUX, 0);
    transpose_cvt_kernel<<<tgridC, thr, 0, stream>>>(WC2k, WCkt, DFFC, DD, KAUX, DFFC);
    transpose_cvt_kernel<<<tgridC, thr, 0, stream>>>(WC1v, WCvt, DFFC, DD, KAUX, 0);
    transpose_cvt_kernel<<<tgridC, thr, 0, stream>>>(WC2v, WCvt, DFFC, DD, KAUX, DFFC);

    // 2) fused modulated projections (WMMA GEMMs)
    dim3 grid(cdiv(DD, BN), MROWS / BM);  // (22, 72)
    gemm_mod_kernel<<<grid, thr, 0, stream>>>(xb, Wqt, bq, Hcat, WCqt, bC1q, bC2q, qb);
    gemm_mod_kernel<<<grid, thr, 0, stream>>>(xb, Wkt, bk, Hcat, WCkt, bC1k, bC2k, kb);
    gemm_mod_kernel<<<grid, thr, 0, stream>>>(xb, Wvt, bv, Hcat, WCvt, bC1v, bC2v, vb);

    // 3) attention (9x9 per batch) + softmax
    attn_scores_kernel<<<MROWS, thr, 0, stream>>>(qb, kb, Aws);

    // 4) attn@V + residual + LayerNorm
    attn_out_ln_kernel<<<MROWS, thr, 0, stream>>>(state, vb, Aws, gamma, beta, out);
}
